// CrossAAttn_adalora_sym_m_81432579932640
// MI455X (gfx1250) — compile-verified, matrix operand order checked by reading, not run on hardware
//
#include <hip/hip_runtime.h>
#include <hip/hip_bf16.h>
#include <stdint.h>

#ifndef __has_builtin
#define __has_builtin(x) 0
#endif

#if __has_builtin(__builtin_amdgcn_tensor_load_to_lds)
#define HAVE_TDM 1
#else
#define HAVE_TDM 0
#endif

typedef __bf16 v16bf __attribute__((ext_vector_type(16)));
typedef float  v8f   __attribute__((ext_vector_type(8)));
typedef unsigned int v4u  __attribute__((ext_vector_type(4)));
typedef int          v8i  __attribute__((ext_vector_type(8)));
typedef int          v4i  __attribute__((ext_vector_type(4)));

// ---------- bf16 helpers: hardware v_cvt on device ----------
__device__ __forceinline__ unsigned short f2bf(float f) {
#if defined(__HIP_DEVICE_COMPILE__)
  union { __bf16 h; unsigned short u; } c;
  c.h = (__bf16)f;               // lowers to v_cvt_pk_bf16_f32 / v_cvt
  return c.u;
#else
  union { float f; unsigned u; } c; c.f = f;
  unsigned r = c.u + 0x7FFFu + ((c.u >> 16) & 1u);
  return (unsigned short)(r >> 16);
#endif
}
__device__ __forceinline__ float bf2f(unsigned short h) {
#if defined(__HIP_DEVICE_COMPILE__)
  union { unsigned short u; __bf16 h; } c; c.u = h;
  return (float)c.h;
#else
  union { unsigned u; float f; } c; c.u = ((unsigned)h) << 16;
  return c.f;
#endif
}

// WMMA operand fragment: 16 bf16 per lane (8 VGPRs)
union FragB {
  v16bf v;
  unsigned short u16[16];
  uint4 q[2];
};

__device__ __forceinline__ v8f wmma_bf16(const FragB& a, const FragB& b, v8f c) {
  return __builtin_amdgcn_wmma_f32_16x16x32_bf16(false, a.v, false, b.v, (short)0, c,
                                                 false, false);
}

// K-index map for CDNA5 16-bit A/B fragments:
// lane<16 : elem e<8 -> K=e,     e>=8 -> K=e+8
// lane>=16: elem e<8 -> K=e+8,   e>=8 -> K=e+16
__device__ __forceinline__ int kmap(int half, int e) {
  return e + ((e >> 3) << 3) + half * 8;
}

#define NPIX 4096          // H*W
#define MTOT 16384         // B*H*W
#define CDIM 256

// ---------------------------------------------------------------------------
// Weight prep: W_eff = bn_scale * (W + 2 * B @ (A*E)),  bias = b - scale*m
// ---------------------------------------------------------------------------
__global__ void prep_w1x1(const float* __restrict__ w,  const float* __restrict__ A,
                          const float* __restrict__ Bm, const float* __restrict__ E,
                          const float* __restrict__ g,  const float* __restrict__ bb,
                          const float* __restrict__ m,  const float* __restrict__ vv,
                          unsigned short* __restrict__ Wout, float* __restrict__ biasOut,
                          int c1, int c2) {
  int idx = blockIdx.x * blockDim.x + threadIdx.x;
  if (idx >= c1 * c2) return;
  int n = idx / c1, k = idx - n * c1;
  float dw = 0.f;
#pragma unroll
  for (int r = 0; r < 8; r++) dw += Bm[n * 8 + r] * E[r] * A[r * c1 + k];
  float s = g[n] * rsqrtf(vv[n] + 1e-5f);
  Wout[(size_t)n * c1 + k] = f2bf((w[idx] + 2.0f * dw) * s);
  if (k == 0) biasOut[n] = bb[n] - s * m[n];
}

__global__ void prep_pe(const float* __restrict__ w,  const float* __restrict__ A,
                        const float* __restrict__ Bm, const float* __restrict__ E,
                        const float* __restrict__ g,  const float* __restrict__ bb,
                        const float* __restrict__ m,  const float* __restrict__ vv,
                        float* __restrict__ Wout, float* __restrict__ biasOut) {
  int idx = blockIdx.x * blockDim.x + threadIdx.x;
  if (idx >= 256 * 49) return;
  int c = idx / 49;
  int rb = idx / 1792, col = idx - rb * 1792;
  float dw = 0.f;
#pragma unroll
  for (int r = 0; r < 8; r++) dw += Bm[rb * 8 + r] * E[r] * A[r * 1792 + col];
  float s = g[c] * rsqrtf(vv[c] + 1e-5f);
  Wout[idx] = (w[idx] + 2.0f * dw) * s;
  if (idx % 49 == 0) biasOut[c] = bb[c] - s * m[c];
}

// ---------------------------------------------------------------------------
// Fused GEMM:  C[m, n] = sum_k A[m,k] * W[n,k] + bias[n]
//   mode 0 (Q):    A = NCHW f32 input (double-buffered TDM -> LDS)
//   mode 1 (KV):   same A path, Nc=512; route to Kmat / Vt (channel-major)
//   mode 2 (PROJ): A = X f32 row-major; out -> d_out NCHW f32
// One wave per block; 16x64 output tile; fully unrolled K (32 WMMAs).
// ---------------------------------------------------------------------------
__global__ __launch_bounds__(32) void gemm_kernel(
    const float* __restrict__ inA_rgb, const float* __restrict__ inA_ir,
    const unsigned short* __restrict__ W, const float* __restrict__ bias,
    unsigned short* __restrict__ outQ, unsigned short* __restrict__ outK,
    unsigned short* __restrict__ outVt, float* __restrict__ outProj,
    int Nc, int mode) {
  const int mt   = blockIdx.x;
  const int cg   = blockIdx.y;
  const int mod  = blockIdx.z;
  const int lane = threadIdx.x;
  const int lm = lane & 15, half = lane >> 4;
  const int nbase = cg * 64;
  const int mrow0 = mt * 16;
  const int b_img = mrow0 >> 12;
  const int n_img = mrow0 & (NPIX - 1);

  const float* A = (mode == 2) ? (inA_rgb + (size_t)mod * MTOT * CDIM)
                               : (mod ? inA_ir : inA_rgb);
  const unsigned short* Wm = W + (size_t)mod * Nc * CDIM;
  const float* bm = bias + (size_t)mod * Nc;

  __shared__ float Atile[2][32 * 16];  // double-buffered [k][m] f32 staging

  v8f zero8 = {0, 0, 0, 0, 0, 0, 0, 0};
  v8f acc[4];
#pragma unroll
  for (int t = 0; t < 4; t++) acc[t] = zero8;

#if HAVE_TDM
  auto tdm_issue = [&](int it) {
    // 2D tile: tile_dim0 = 16 pixels (contiguous), tile_dim1 = 32 channels
    const float* src = A + ((size_t)b_img * CDIM + it * 32) * NPIX + n_img;
    unsigned long long ga = (unsigned long long)(const void*)src;
    unsigned lds = (unsigned)(uintptr_t)(void*)&Atile[it & 1][0];
    v4u g0 = {1u, lds, (unsigned)ga, (unsigned)(ga >> 32) | (2u << 30)};
    v8i g1 = {(int)(2u << 16),                               // data_size = 4B
              (int)(((unsigned)NPIX) << 16),                 // tensor_dim0 lo16
              (int)((((unsigned)NPIX) >> 16) | (256u << 16)),// d0 hi | d1 lo
              (int)((256u >> 16) | (16u << 16)),             // d1 hi | tile_dim0=16
              32,                                            // tile_dim1=32
              NPIX,                                          // dim0 stride
              0, 0};
    v4i g2 = {0, 0, 0, 0};
    v4i g3 = {0, 0, 0, 0};
#if __clang_major__ >= 23
    v8i g4 = {0, 0, 0, 0, 0, 0, 0, 0};
    __builtin_amdgcn_tensor_load_to_lds(g0, g1, g2, g3, g4, 0);
#else
    __builtin_amdgcn_tensor_load_to_lds(g0, g1, g2, g3, 0);
#endif
  };
#endif

  auto do_tiles = [&](const FragB& afrag, int kb) {
#pragma unroll
    for (int t = 0; t < 4; t++) {
      FragB bfrag;
      const unsigned short* wp =
          Wm + (size_t)(nbase + t * 16 + lm) * CDIM + kb + half * 8;
      bfrag.q[0] = *(const uint4*)(wp);
      bfrag.q[1] = *(const uint4*)(wp + 16);
      acc[t] = wmma_bf16(afrag, bfrag, acc[t]);
    }
  };

  if (mode == 2) {
#pragma unroll
    for (int it = 0; it < 8; it++) {
      FragB afrag;
      const float* ap = A + (size_t)(mrow0 + lm) * CDIM + it * 32;
#pragma unroll
      for (int e = 0; e < 16; e++) afrag.u16[e] = f2bf(ap[kmap(half, e)]);
      do_tiles(afrag, it * 32);
    }
  } else {
#if HAVE_TDM
    tdm_issue(0);
#pragma unroll
    for (int it = 0; it < 8; it++) {
      if (it + 1 < 8) {
        tdm_issue(it + 1);
        __builtin_amdgcn_s_wait_tensorcnt(1);  // buffer 'it' resident; next in flight
      } else {
        __builtin_amdgcn_s_wait_tensorcnt(0);
      }
      FragB afrag;
#pragma unroll
      for (int e = 0; e < 16; e++)
        afrag.u16[e] = f2bf(Atile[it & 1][kmap(half, e) * 16 + lm]);
      do_tiles(afrag, it * 32);
    }
#else
    for (int it = 0; it < 8; it++) {
      const float* src = A + ((size_t)b_img * CDIM + it * 32) * NPIX + n_img;
      for (int i = lane; i < 32 * 16; i += 32) {
        int kk = i >> 4, mm = i & 15;
        Atile[0][kk * 16 + mm] = src[(size_t)kk * NPIX + mm];
      }
      __syncthreads();
      FragB afrag;
#pragma unroll
      for (int e = 0; e < 16; e++)
        afrag.u16[e] = f2bf(Atile[0][kmap(half, e) * 16 + lm]);
      __syncthreads();
      do_tiles(afrag, it * 32);
    }
#endif
  }

  // Epilogue (C layout: lane<16 -> rows i, lanes>=16 -> rows i+8; col = lane%16)
#pragma unroll
  for (int t = 0; t < 4; t++) {
#pragma unroll
    for (int i = 0; i < 8; i++) {
      int Mr = i + half * 8;
      int mrow = mrow0 + Mr;
      int c = nbase + t * 16 + lm;
      float val = acc[t][i] + bm[c];
      if (mode == 0) {
        outQ[(size_t)mod * MTOT * CDIM + (size_t)mrow * CDIM + c] = f2bf(val);
      } else if (mode == 1) {
        int h = c >> 6, r = c & 63;
        if (r < 32)
          outK[(size_t)mod * MTOT * CDIM + (size_t)mrow * CDIM + h * 32 + r] = f2bf(val);
        else
          outVt[(size_t)mod * CDIM * MTOT + (size_t)(h * 32 + r - 32) * MTOT + mrow] =
              f2bf(val);
      } else {
        int bb2 = mrow >> 12, nn = mrow & (NPIX - 1);
        outProj[((size_t)(mod * 4 + bb2) * CDIM + c) * NPIX + nn] = val;
      }
    }
  }
}

// ---------------------------------------------------------------------------
// Flash attention: one wave per 16 q-rows of one (mod, window, head).
// ---------------------------------------------------------------------------
__global__ __launch_bounds__(32) void attn_kernel(
    const unsigned short* __restrict__ Qmat, const unsigned short* __restrict__ Kmat,
    const unsigned short* __restrict__ Vt, float* __restrict__ X) {
  const int qt = blockIdx.x;       // 0..63 q-tiles
  const int bh = blockIdx.y;       // 0..127 = (window-batch, head)
  const int mod = blockIdx.z;
  const int ba = bh >> 3, h = bh & 7;
  const int b = ba >> 2, win = ba & 3;
  const int lane = threadIdx.x, lm = lane & 15, half = lane >> 4;
  const size_t modOff = (size_t)mod * MTOT * CDIM;
  const int m0 = b * NPIX + win * 1024;
  const int qbase = m0 + qt * 16;
  const float sc = 0.17677669529663687f;  // 1/sqrt(32)

  __shared__ unsigned short Pt[16 * 32];  // [m][key] bf16

  FragB qf;
  {
    const unsigned short* qp =
        Qmat + modOff + (size_t)(qbase + lm) * CDIM + h * 32 + half * 8;
    qf.q[0] = *(const uint4*)qp;
    qf.q[1] = *(const uint4*)(qp + 16);
  }

  v8f zero8 = {0, 0, 0, 0, 0, 0, 0, 0};
  v8f o0 = zero8, o1 = zero8;
  float mrun[8], lrun[8];
#pragma unroll
  for (int i = 0; i < 8; i++) { mrun[i] = -1e30f; lrun[i] = 0.f; }

  for (int j = 0; j < 32; j++) {
    const int kb = m0 + j * 32;
    FragB kf0, kf1;
    {
      const unsigned short* kp =
          Kmat + modOff + (size_t)(kb + lm) * CDIM + h * 32 + half * 8;
      kf0.q[0] = *(const uint4*)kp;
      kf0.q[1] = *(const uint4*)(kp + 16);
      kp += (size_t)16 * CDIM;
      kf1.q[0] = *(const uint4*)kp;
      kf1.q[1] = *(const uint4*)(kp + 16);
    }
    v8f s0 = wmma_bf16(qf, kf0, zero8);
    v8f s1 = wmma_bf16(qf, kf1, zero8);

    float p0[8], p1[8];
#pragma unroll
    for (int i = 0; i < 8; i++) {
      float rm = fmaxf(s0[i], s1[i]);
#pragma unroll
      for (int off = 1; off < 16; off <<= 1) rm = fmaxf(rm, __shfl_xor(rm, off, 32));
      float mn = fmaxf(mrun[i], rm * sc);
      float alpha = __expf(mrun[i] - mn);
      float e0 = __expf(s0[i] * sc - mn);
      float e1 = __expf(s1[i] * sc - mn);
      float rs = e0 + e1;
#pragma unroll
      for (int off = 1; off < 16; off <<= 1) rs += __shfl_xor(rs, off, 32);
      lrun[i] = lrun[i] * alpha + rs;
      mrun[i] = mn;
      o0[i] *= alpha;
      o1[i] *= alpha;
      p0[i] = e0;
      p1[i] = e1;
    }

    // P (C layout) -> LDS [m][key]
#pragma unroll
    for (int i = 0; i < 8; i++) {
      int Mr = i + half * 8;
      Pt[Mr * 32 + lm] = f2bf(p0[i]);
      Pt[Mr * 32 + 16 + lm] = f2bf(p1[i]);
    }
    __syncthreads();
    FragB pf;
#pragma unroll
    for (int e = 0; e < 16; e++) pf.u16[e] = Pt[lm * 32 + kmap(half, e)];
    __syncthreads();

    FragB vf0, vf1;
    {
      const unsigned short* vp =
          Vt + modOff + (size_t)(h * 32 + lm) * MTOT + kb + half * 8;
      vf0.q[0] = *(const uint4*)vp;
      vf0.q[1] = *(const uint4*)(vp + 16);
      vp += (size_t)16 * MTOT;
      vf1.q[0] = *(const uint4*)vp;
      vf1.q[1] = *(const uint4*)(vp + 16);
    }
    o0 = wmma_bf16(pf, vf0, o0);
    o1 = wmma_bf16(pf, vf1, o1);
  }

#pragma unroll
  for (int i = 0; i < 8; i++) {
    int Mr = i + half * 8;
    float inv = 1.0f / lrun[i];
    size_t base = modOff + (size_t)(qbase + Mr) * CDIM + h * 32;
    X[base + lm] = o0[i] * inv;
    X[base + 16 + lm] = o1[i] * inv;
  }
}

// ---------------------------------------------------------------------------
// Depthwise 7x7 conv on V (bf16, channel-major), accumulate into X (f32).
// ---------------------------------------------------------------------------
__global__ void pe_kernel(const unsigned short* __restrict__ Vt,
                          const float* __restrict__ Wpe, const float* __restrict__ bpe,
                          float* __restrict__ X) {
  int idx = blockIdx.x * blockDim.x + threadIdx.x;
  int n = idx & (NPIX - 1);
  int t = idx >> 12;
  int c = t & 255; t >>= 8;
  int b = t & 3;
  int mod = t >> 2;
  if (mod >= 2) return;
  int y = n >> 6, x = n & 63;
  const unsigned short* v = Vt + ((size_t)mod * CDIM + c) * MTOT + (size_t)b * NPIX;
  const float* w = Wpe + ((size_t)mod * CDIM + c) * 49;
  float acc = bpe[mod * CDIM + c];
#pragma unroll
  for (int i = 0; i < 7; i++) {
    int yy = y + i - 3;
    if ((unsigned)yy >= 64u) continue;
#pragma unroll
    for (int jj = 0; jj < 7; jj++) {
      int xx = x + jj - 3;
      if ((unsigned)xx >= 64u) continue;
      acc += w[i * 7 + jj] * bf2f(v[yy * 64 + xx]);
    }
  }
  X[((size_t)mod * MTOT + (size_t)b * NPIX + n) * CDIM + c] += acc;
}

// ---------------------------------------------------------------------------
extern "C" void kernel_launch(void* const* d_in, const int* in_sizes, int n_in,
                              void* d_out, int out_size, void* d_ws, size_t ws_size,
                              hipStream_t stream) {
  // params flattened in sorted-key pytree order: blocks {kv, pe, proj, q},
  // keys {A_ir, A_rgb, B_ir, B_rgb, E, b_ir, b_rgb, g_ir, g_rgb, m_ir, m_rgb, v_ir, v_rgb, w}
  enum { A_ir = 0, A_rgb, B_ir, B_rgb, Ek, b_ir, b_rgb, g_ir, g_rgb, m_ir, m_rgb, v_ir, v_rgb, Wk };
  enum { BKV = 0, BPE = 1, BPROJ = 2, BQ = 3 };
  auto P = [&](int block, int key) { return (const float*)d_in[4 + block * 14 + key]; };

  char* ws = (char*)d_ws;
  size_t off = 0;
  auto alloc = [&](size_t bytes) {
    void* p = ws + off;
    off += (bytes + 255) & ~(size_t)255;
    return p;
  };
  unsigned short* Wq    = (unsigned short*)alloc((size_t)2 * 256 * 256 * 2);
  unsigned short* Wkv   = (unsigned short*)alloc((size_t)2 * 512 * 256 * 2);
  unsigned short* Wproj = (unsigned short*)alloc((size_t)2 * 256 * 256 * 2);
  float* bq    = (float*)alloc(2 * 256 * 4);
  float* bkv   = (float*)alloc(2 * 512 * 4);
  float* bproj = (float*)alloc(2 * 256 * 4);
  float* Wpe   = (float*)alloc((size_t)2 * 256 * 49 * 4);
  float* bpe   = (float*)alloc(2 * 256 * 4);
  unsigned short* Qm = (unsigned short*)alloc((size_t)2 * MTOT * CDIM * 2);
  unsigned short* Km = (unsigned short*)alloc((size_t)2 * MTOT * CDIM * 2);
  unsigned short* Vt = (unsigned short*)alloc((size_t)2 * MTOT * CDIM * 2);
  float* X = (float*)alloc((size_t)2 * MTOT * CDIM * 4);

  for (int mod = 0; mod < 2; mod++) {
    int Ai = mod ? A_ir : A_rgb, Bi = mod ? B_ir : B_rgb;
    int gi = mod ? g_ir : g_rgb, bi = mod ? b_ir : b_rgb;
    int mi = mod ? m_ir : m_rgb, vi = mod ? v_ir : v_rgb;
    prep_w1x1<<<(256 * 256 + 255) / 256, 256, 0, stream>>>(
        P(BQ, Wk), P(BQ, Ai), P(BQ, Bi), P(BQ, Ek), P(BQ, gi), P(BQ, bi), P(BQ, mi),
        P(BQ, vi), Wq + (size_t)mod * 256 * 256, bq + mod * 256, 256, 256);
    prep_w1x1<<<(512 * 256 + 255) / 256, 256, 0, stream>>>(
        P(BKV, Wk), P(BKV, Ai), P(BKV, Bi), P(BKV, Ek), P(BKV, gi), P(BKV, bi),
        P(BKV, mi), P(BKV, vi), Wkv + (size_t)mod * 512 * 256, bkv + mod * 512, 256, 512);
    prep_w1x1<<<(256 * 256 + 255) / 256, 256, 0, stream>>>(
        P(BPROJ, Wk), P(BPROJ, Ai), P(BPROJ, Bi), P(BPROJ, Ek), P(BPROJ, gi),
        P(BPROJ, bi), P(BPROJ, mi), P(BPROJ, vi), Wproj + (size_t)mod * 256 * 256,
        bproj + mod * 256, 256, 256);
    prep_pe<<<(256 * 49 + 255) / 256, 256, 0, stream>>>(
        P(BPE, Wk), P(BPE, Ai), P(BPE, Bi), P(BPE, Ek), P(BPE, gi), P(BPE, bi),
        P(BPE, mi), P(BPE, vi), Wpe + (size_t)mod * 256 * 49, bpe + mod * 256);
  }

  gemm_kernel<<<dim3(MTOT / 16, 4, 2), 32, 0, stream>>>(
      (const float*)d_in[0], (const float*)d_in[1], Wq, bq, Qm, nullptr, nullptr,
      nullptr, 256, 0);
  gemm_kernel<<<dim3(MTOT / 16, 8, 2), 32, 0, stream>>>(
      (const float*)d_in[2], (const float*)d_in[3], Wkv, bkv, nullptr, Km, Vt,
      nullptr, 512, 1);
  attn_kernel<<<dim3(64, 128, 2), 32, 0, stream>>>(Qm, Km, Vt, X);
  pe_kernel<<<(2 * 4 * CDIM * NPIX) / 256, 256, 0, stream>>>(Vt, Wpe, bpe, X);
  gemm_kernel<<<dim3(MTOT / 16, 4, 2), 32, 0, stream>>>(
      X, nullptr, Wproj, bproj, nullptr, nullptr, nullptr, (float*)d_out, 256, 2);
}
